// TreeLSTM_78383153152087
// MI455X (gfx1250) — matrix operand order repaired and checked by reading, not compiled, not run on hardware
//
#include <hip/hip_runtime.h>

// Problem dimensions (fixed by the reference).
#define Tn  256
#define Bn  512
#define INn 128
#define Mn  256
#define BFn 2
#define NBLK 64   // persistent grid: 64 blocks x 256 threads = 512 waves (one per 16x16 tile)

typedef __attribute__((ext_vector_type(16))) __bf16 v16bf;
typedef __attribute__((ext_vector_type(8)))  float  v8f;
typedef __attribute__((ext_vector_type(4)))  unsigned int v4u;

union FragU { v4u q[2]; v16bf v; };

__device__ __forceinline__ unsigned short f2bf(float f) {
  unsigned int u = __float_as_uint(f);
  unsigned int r = (u + 0x7FFFu + ((u >> 16) & 1u)) >> 16;  // round-to-nearest-even
  return (unsigned short)r;
}

// A-matrix 16x32 bf16 fragment (wave32): lane L(0-15)=row L holds K {kBase..kBase+7, kBase+16..kBase+23},
// kBase = (lane>>4)*8. rowChunk points at this lane's row at the 32-wide K chunk start (global memory).
__device__ __forceinline__ v16bf load_a_frag(const unsigned short* rowChunk, int aKBase) {
  FragU f;
  f.q[0] = *(const v4u*)(rowChunk + aKBase);
  f.q[1] = *(const v4u*)(rowChunk + aKBase + 16);
  return f.v;
}

__device__ __forceinline__ v16bf zero_frag() {
  FragU f;
  v4u z = {0u, 0u, 0u, 0u};
  f.q[0] = z; f.q[1] = z;
  return f.v;
}

// B-matrix 32x16 fragment served from LDS: lane n(0-15)=col n holds K {0..7,16..23}, lane n+16 holds K {8..15,24..31}.
__device__ __forceinline__ v16bf lds_b_frag(const unsigned short* p) {
  FragU f;
  f.q[0] = *(const v4u*)(p);
  f.q[1] = *(const v4u*)(p + 16);
  return f.v;
}

__device__ __forceinline__ float sigmoidf_(float x) { return 1.0f / (1.0f + __expf(-x)); }
__device__ __forceinline__ float tanhf_(float x)    { return 2.0f / (1.0f + __expf(-2.0f * x)) - 1.0f; }

// ---------------------------------------------------------------------------
// f32 -> bf16 elementwise conversion
// ---------------------------------------------------------------------------
__global__ void f32_to_bf16_kernel(const float* __restrict__ src,
                                   unsigned short* __restrict__ dst, int n) {
  int i = blockIdx.x * blockDim.x + threadIdx.x;
  if (i < n) dst[i] = f2bf(src[i]);
}

// ---------------------------------------------------------------------------
// Replay the stack machine per tree: emit child gather indices (-1 = masked)
// and the root index. One thread per batch column.
// ---------------------------------------------------------------------------
__global__ void precompute_kernel(const int* __restrict__ arities,
                                  int* __restrict__ childIdx,
                                  int* __restrict__ rootIdx,
                                  int* __restrict__ stack) {
  int b = blockIdx.x * blockDim.x + threadIdx.x;
  if (b >= Bn) return;
  for (int p = 0; p < Tn + BFn; ++p) stack[p * Bn + b] = 0;
  int sp = BFn - 1;
  for (int t = 0; t < Tn; ++t) {
    int ar = arities[t * Bn + b];
    #pragma unroll
    for (int k = 0; k < BFn; ++k) {
      int pos = sp - k;
      int idx = (pos >= 0 && pos < Tn + BFn) ? stack[pos * Bn + b] : 0;
      childIdx[((size_t)t * Bn + b) * BFn + k] = (ar > k) ? idx : -1;
    }
    sp = sp + 1 - (ar < 0 ? -ar : ar);
    if (ar != -1 && sp >= 0 && sp < Tn + BFn) stack[sp * Bn + b] = t;
  }
  rootIdx[b] = (sp >= 0 && sp < Tn + BFn) ? stack[sp * Bn + b] : 0;
}

// ---------------------------------------------------------------------------
// Persistent Tree-LSTM scan. Grid = 64 blocks x 256 threads (8 waves).
// Block -> (batch quarter, feature tile); all 8 waves share one 16-feature
// column tile, whose step-invariant weights live in LDS for the whole scan.
// Grid-wide release/acquire barrier between timesteps.
// ---------------------------------------------------------------------------
__global__ void __launch_bounds__(256)
tree_scan_kernel(const unsigned short* __restrict__ Xb,    // (T,B,IN) bf16
                 const unsigned short* __restrict__ Wib,   // (M,IN)   bf16
                 const unsigned short* __restrict__ Wob,
                 const unsigned short* __restrict__ Wub,
                 const unsigned short* __restrict__ Uib,   // (BF,M,M) bf16
                 const unsigned short* __restrict__ Uob,
                 const unsigned short* __restrict__ Uub,
                 const float* __restrict__ bi,
                 const float* __restrict__ bo,
                 const float* __restrict__ bu,
                 const int* __restrict__ childIdx,         // (T,B,BF)
                 const int* __restrict__ rootIdx,          // (B)
                 unsigned short* __restrict__ Hb,          // (T,B,M) bf16 state
                 float* __restrict__ Hf,                   // (T,B,M) f32 = d_out + B*M
                 float* __restrict__ outRoot,              // (B,M)
                 int* __restrict__ counter)                // grid barrier
{
  __shared__ __align__(16) unsigned short sW[3][16][INn];        // 12 KB
  __shared__ __align__(16) unsigned short sU[3][BFn][16][Mn];    // 48 KB

  const int fTile = blockIdx.x & 15;          // feature tile (shared by block)
  const int quart = blockIdx.x >> 4;          // batch quarter
  const int waveId = threadIdx.x >> 5;
  const int lane   = threadIdx.x & 31;

  const int mBase = fTile << 4;
  const int bBase = ((quart << 3) + waveId) << 4;

  const int laneN  = lane & 15;
  const int kHalf  = lane >> 4;
  const int col    = mBase + laneN;           // output feature (B/C/D column)
  const int aRow   = bBase + laneN;           // A-matrix row (batch)
  const int aKBase = kHalf << 3;              // 0 or 8
  const int bKOff  = kHalf << 3;              // 0 or 8

  // ---- stage step-invariant weight tiles into LDS (once for all 256 steps) ----
  {
    const unsigned short* Wsrc[3] = {Wib, Wob, Wub};
    const unsigned short* Usrc[3] = {Uib, Uob, Uub};
    for (int g = 0; g < 3; ++g) {
      for (int i = threadIdx.x; i < 16 * (INn / 8); i += blockDim.x) {
        int r = i / (INn / 8), c = (i % (INn / 8)) * 8;
        *(v4u*)&sW[g][r][c] = *(const v4u*)(Wsrc[g] + (size_t)(mBase + r) * INn + c);
      }
      for (int k = 0; k < BFn; ++k) {
        for (int i = threadIdx.x; i < 16 * (Mn / 8); i += blockDim.x) {
          int r = i / (Mn / 8), c = (i % (Mn / 8)) * 8;
          *(v4u*)&sU[g][k][r][c] =
              *(const v4u*)(Usrc[g] + ((size_t)k * Mn + (mBase + r)) * Mn + c);
        }
      }
    }
  }
  __syncthreads();

  const float vi = bi[col], vo = bo[col], vu = bu[col];

  for (int t = 0; t < Tn; ++t) {
    v8f acc_i, acc_o, acc_u;
    #pragma unroll
    for (int j = 0; j < 8; ++j) { acc_i[j] = vi; acc_o[j] = vo; acc_u[j] = vu; }

    // Prefetch next timestep's x row while we compute (global_prefetch_b8).
    if (t + 1 < Tn)
      __builtin_prefetch(Xb + ((size_t)(t + 1) * Bn + aRow) * INn, 0, 1);

    // ---- input projection: X[t] (BxIN) @ W^T ----
    const unsigned short* xrow = Xb + ((size_t)t * Bn + aRow) * INn;
    #pragma unroll
    for (int kc = 0; kc < INn; kc += 32) {
      v16bf a  = load_a_frag(xrow + kc, aKBase);
      v16bf fi = lds_b_frag(&sW[0][laneN][kc + bKOff]);
      v16bf fo = lds_b_frag(&sW[1][laneN][kc + bKOff]);
      v16bf fu = lds_b_frag(&sW[2][laneN][kc + bKOff]);
      acc_i = __builtin_amdgcn_wmma_f32_16x16x32_bf16(false, a, false, fi, (short)0, acc_i, false, false);
      acc_o = __builtin_amdgcn_wmma_f32_16x16x32_bf16(false, a, false, fo, (short)0, acc_o, false, false);
      acc_u = __builtin_amdgcn_wmma_f32_16x16x32_bf16(false, a, false, fu, (short)0, acc_u, false, false);
    }

    // ---- child projections: h(child_k) (BxM) @ U_k^T ----
    #pragma unroll
    for (int k = 0; k < BFn; ++k) {
      const int idx = childIdx[((size_t)t * Bn + aRow) * BFn + k];
      const unsigned short* hrow = Hb + ((size_t)(idx < 0 ? 0 : idx) * Bn + aRow) * Mn;
      #pragma unroll
      for (int kc = 0; kc < Mn; kc += 32) {
        v16bf a = zero_frag();
        if (idx >= 0) a = load_a_frag(hrow + kc, aKBase);  // reconverges before WMMA
        v16bf fi = lds_b_frag(&sU[0][k][laneN][kc + bKOff]);
        v16bf fo = lds_b_frag(&sU[1][k][laneN][kc + bKOff]);
        v16bf fu = lds_b_frag(&sU[2][k][laneN][kc + bKOff]);
        acc_i = __builtin_amdgcn_wmma_f32_16x16x32_bf16(false, a, false, fi, (short)0, acc_i, false, false);
        acc_o = __builtin_amdgcn_wmma_f32_16x16x32_bf16(false, a, false, fo, (short)0, acc_o, false, false);
        acc_u = __builtin_amdgcn_wmma_f32_16x16x32_bf16(false, a, false, fu, (short)0, acc_u, false, false);
      }
    }

    // ---- nonlinearities + store h (D layout: VGPR j -> row kHalf*8+j, col laneN) ----
    const int outRowBase = bBase + (kHalf << 3);
    #pragma unroll
    for (int j = 0; j < 8; ++j) {
      float c = sigmoidf_(acc_i[j]) * tanhf_(acc_u[j]);
      float h = sigmoidf_(acc_o[j]) * tanhf_(c);
      size_t o = ((size_t)t * Bn + (outRowBase + j)) * Mn + col;
      Hf[o] = h;
      Hb[o] = f2bf(h);
    }

    // ---- grid-wide barrier: release this step's h, acquire everyone else's ----
    __syncthreads();
    if (threadIdx.x == 0) {
      __hip_atomic_fetch_add(counter, 1, __ATOMIC_RELEASE, __HIP_MEMORY_SCOPE_AGENT);
      const int target = (t + 1) * (int)gridDim.x;
      while (__hip_atomic_load(counter, __ATOMIC_ACQUIRE, __HIP_MEMORY_SCOPE_AGENT) < target) {
        __builtin_amdgcn_s_sleep(2);
      }
    }
    __syncthreads();
    __threadfence();   // agent-scope acquire for all threads before reading peers' h
  }

  // ---- fused root gather (all h visible after final barrier) ----
  for (int i = blockIdx.x * blockDim.x + threadIdx.x; i < Bn * Mn;
       i += (int)gridDim.x * blockDim.x) {
    int b = i / Mn, m = i % Mn;
    outRoot[i] = Hf[((size_t)rootIdx[b] * Bn + b) * Mn + m];
  }
}

// ---------------------------------------------------------------------------
extern "C" void kernel_launch(void* const* d_in, const int* in_sizes, int n_in,
                              void* d_out, int out_size, void* d_ws, size_t ws_size,
                              hipStream_t stream) {
  (void)in_sizes; (void)n_in; (void)out_size; (void)ws_size;

  const float* X  = (const float*)d_in[0];
  const int*   ar = (const int*)  d_in[1];
  const float* Wi = (const float*)d_in[2];
  const float* bi = (const float*)d_in[3];
  const float* Wo = (const float*)d_in[4];
  const float* bo = (const float*)d_in[5];
  const float* Wu = (const float*)d_in[6];
  const float* bu = (const float*)d_in[7];
  // d_in[8]=Wf, d_in[9]=bf_b, d_in[13]=Uf: dead in the reference
  const float* Ui = (const float*)d_in[10];
  const float* Uo = (const float*)d_in[11];
  const float* Uu = (const float*)d_in[12];

  float* outRoot = (float*)d_out;                 // (B, M) root h
  float* outMem  = outRoot + (size_t)Bn * Mn;     // (T, B, M) h history

  char* ws = (char*)d_ws;
  size_t off = 0;
  int* counter  = (int*)(ws + off); off += 16;    // grid barrier (16B keeps alignment)
  int* childIdx = (int*)(ws + off); off += (size_t)Tn * Bn * BFn * sizeof(int);
  int* rootIdx  = (int*)(ws + off); off += (size_t)Bn * sizeof(int);
  int* stack    = (int*)(ws + off); off += (size_t)(Tn + BFn) * Bn * sizeof(int);
  unsigned short* Xb  = (unsigned short*)(ws + off); off += (size_t)Tn * Bn * INn * 2;
  unsigned short* Wib = (unsigned short*)(ws + off); off += (size_t)Mn * INn * 2;
  unsigned short* Wob = (unsigned short*)(ws + off); off += (size_t)Mn * INn * 2;
  unsigned short* Wub = (unsigned short*)(ws + off); off += (size_t)Mn * INn * 2;
  unsigned short* Uib = (unsigned short*)(ws + off); off += (size_t)BFn * Mn * Mn * 2;
  unsigned short* Uob = (unsigned short*)(ws + off); off += (size_t)BFn * Mn * Mn * 2;
  unsigned short* Uub = (unsigned short*)(ws + off); off += (size_t)BFn * Mn * Mn * 2;
  unsigned short* Hb  = (unsigned short*)(ws + off); off += (size_t)Tn * Bn * Mn * 2;

  // Deterministic state each call (capture-safe async memsets).
  hipMemsetAsync(counter, 0, 16, stream);
  hipMemsetAsync(Hb, 0, (size_t)Tn * Bn * Mn * 2, stream);

  auto conv = [&](const float* s, unsigned short* d, int n) {
    f32_to_bf16_kernel<<<dim3((n + 255) / 256), dim3(256), 0, stream>>>(s, d, n);
  };
  conv(X,  Xb,  Tn * Bn * INn);
  conv(Wi, Wib, Mn * INn);
  conv(Wo, Wob, Mn * INn);
  conv(Wu, Wub, Mn * INn);
  conv(Ui, Uib, BFn * Mn * Mn);
  conv(Uo, Uob, BFn * Mn * Mn);
  conv(Uu, Uub, BFn * Mn * Mn);

  precompute_kernel<<<dim3((Bn + 255) / 256), dim3(256), 0, stream>>>(ar, childIdx, rootIdx, stack);

  // Single persistent kernel: 64 blocks (trivially co-resident), 256 steps inside.
  tree_scan_kernel<<<dim3(NBLK), dim3(256), 0, stream>>>(
      Xb, Wib, Wob, Wub, Uib, Uob, Uub, bi, bo, bu,
      childIdx, rootIdx, Hb, outMem, outRoot, counter);
}
